// WormholeTessellationExpert_88493506167408
// MI455X (gfx1250) — matrix operand order repaired and checked by reading, not compiled, not run on hardware
//
#include <hip/hip_runtime.h>

#define BB 4
#define SS 4096
#define DN 1024
#define TN 16
#define TDI 64
#define KW 4

typedef __attribute__((ext_vector_type(8)))  float  v8f;
typedef __attribute__((ext_vector_type(8)))  __bf16 v8bf;
typedef __attribute__((ext_vector_type(16))) __bf16 v16bf;
typedef __attribute__((ext_vector_type(4)))  __bf16 v4bf;

static __device__ __forceinline__ v16bf cat16(v8bf lo, v8bf hi) {
  return __builtin_shufflevector(lo, hi, 0,1,2,3,4,5,6,7,8,9,10,11,12,13,14,15);
}

// ---------------------------------------------------------------------------
// Kernel 1: pre-swizzle W1 [128,320] and W2 [64,128] (row-major, out x in)
// into bf16 WMMA B-fragments.  B (32x16), contiguous k order of this file (A matches): lane L holds N = L&15,
// element e holds K = (L>=16 ? 16 : 0) + e.  Fragment f[kt][nt][lane][e].
// ---------------------------------------------------------------------------
__global__ __launch_bounds__(256) void wte_prep(const float* __restrict__ W1,
                                                const float* __restrict__ W2,
                                                __bf16* __restrict__ w1f,
                                                __bf16* __restrict__ w2f) {
  int idx = blockIdx.x * 256 + threadIdx.x;
  const int total1 = 10 * 8 * 32 * 16;          // kt x nt x lane x e
  if (idx < total1) {
    int e = idx & 15, lane = (idx >> 4) & 31, nt = (idx >> 9) & 7, kt = idx >> 12;
    int n  = nt * 16 + (lane & 15);
    int kk = kt * 32 + ((lane >= 16) ? 16 : 0) + e;
    w1f[idx] = (__bf16)W1[n * 320 + kk];
  } else {
    int i2 = idx - total1;
    if (i2 < 4 * 4 * 32 * 16) {
      int e = i2 & 15, lane = (i2 >> 4) & 31, nt = (i2 >> 9) & 3, kt = i2 >> 11;
      int n  = nt * 16 + (lane & 15);
      int kk = kt * 32 + ((lane >= 16) ? 16 : 0) + e;
      w2f[i2] = (__bf16)W2[n * 128 + kk];
    }
  }
}

// ---------------------------------------------------------------------------
// Kernel 2: LayerNorm + per-block partial sums of xn over 32 tokens.
// Grid 512 blocks (128 per batch), 256 threads, thread owns 4 features.
// partial[block][feature] -> summed deterministically in the route kernel.
// ---------------------------------------------------------------------------
__global__ __launch_bounds__(256) void wte_stats(const float* __restrict__ x,
                                                 const float* __restrict__ gamma,
                                                 const float* __restrict__ beta,
                                                 float* __restrict__ partial) {
  __shared__ float red1[8], red2[8];
  __shared__ float s_mu, s_rs;
  int tid = threadIdx.x, lane = tid & 31, wv = tid >> 5;
  long tok0 = (long)blockIdx.x * 32;
  float4 gv = ((const float4*)gamma)[tid];
  float4 bv = ((const float4*)beta)[tid];
  float a0 = 0.f, a1 = 0.f, a2 = 0.f, a3 = 0.f;
  for (int t = 0; t < 32; ++t) {
    long tok = tok0 + t;
    float4 xv = ((const float4*)(x + tok * DN))[tid];
    float s  = xv.x + xv.y + xv.z + xv.w;
    float s2 = xv.x * xv.x + xv.y * xv.y + xv.z * xv.z + xv.w * xv.w;
#pragma unroll
    for (int off = 16; off > 0; off >>= 1) {
      s  += __shfl_down(s, off, 32);
      s2 += __shfl_down(s2, off, 32);
    }
    if (lane == 0) { red1[wv] = s; red2[wv] = s2; }
    __syncthreads();
    if (tid == 0) {
      float u = 0.f, u2 = 0.f;
#pragma unroll
      for (int w = 0; w < 8; ++w) { u += red1[w]; u2 += red2[w]; }
      float mu = u * (1.0f / DN);
      s_mu = mu;
      s_rs = rsqrtf(u2 * (1.0f / DN) - mu * mu + 1e-5f);
    }
    __syncthreads();
    float mu = s_mu, rs = s_rs;
    a0 += (xv.x - mu) * rs * gv.x + bv.x;
    a1 += (xv.y - mu) * rs * gv.y + bv.y;
    a2 += (xv.z - mu) * rs * gv.z + bv.z;
    a3 += (xv.w - mu) * rs * gv.w + bv.w;
  }
  float* p = partial + (long)blockIdx.x * DN + tid * 4;
  p[0] = a0; p[1] = a1; p[2] = a2; p[3] = a3;
}

// ---------------------------------------------------------------------------
// Kernel 3: routing.  One block.  tile means -> q,k -> l2norm -> scores
// (diag masked) -> top-4 per (b,t) -> routes[B*T*K] ints.
// ---------------------------------------------------------------------------
__global__ __launch_bounds__(256) void wte_route(const float* __restrict__ partial,
                                                 const float* __restrict__ Wq,
                                                 const float* __restrict__ bq,
                                                 const float* __restrict__ Wk,
                                                 const float* __restrict__ bk,
                                                 int* __restrict__ rout) {
  __shared__ float tr_s[BB][DN];
  __shared__ float q_s[BB][TN][TDI];
  __shared__ float k_s[BB][TN][TDI];
  __shared__ float sc_s[BB][TN][TN];
  int tid = threadIdx.x;
  for (int i = tid; i < BB * DN; i += 256) {
    int b = i >> 10, f = i & (DN - 1);
    float s = 0.f;
    for (int blk = 0; blk < SS / 32; ++blk)
      s += partial[((long)(b * (SS / 32) + blk)) * DN + f];
    tr_s[b][f] = s * (1.0f / SS);
  }
  __syncthreads();
  for (int i = tid; i < BB * TN * TDI; i += 256) {
    int b = i >> 10, t = (i >> 6) & 15, o = i & 63;
    const float* tr = &tr_s[b][t * TDI];
    float aq = bq[o], ak = bk[o];
    for (int c = 0; c < TDI; ++c) {
      float v = tr[c];
      aq += v * Wq[o * TDI + c];
      ak += v * Wk[o * TDI + c];
    }
    q_s[b][t][o] = aq;
    k_s[b][t][o] = ak;
  }
  __syncthreads();
  for (int i = tid; i < BB * TN * 2; i += 256) {
    int b = i >> 5, t = (i >> 1) & 15;
    float* v = (i & 1) ? k_s[b][t] : q_s[b][t];
    float n = 0.f;
    for (int c = 0; c < TDI; ++c) n += v[c] * v[c];
    float inv = 1.0f / fmaxf(sqrtf(n), 1e-12f);
    for (int c = 0; c < TDI; ++c) v[c] *= inv;
  }
  __syncthreads();
  for (int i = tid; i < BB * TN * TN; i += 256) {
    int b = i >> 8, t = (i >> 4) & 15, u = i & 15;
    float s = 0.f;
    for (int c = 0; c < TDI; ++c) s += q_s[b][t][c] * k_s[b][u][c];
    sc_s[b][t][u] = (t == u) ? -1e9f : s;
  }
  __syncthreads();
  if (tid < BB * TN) {
    int b = tid >> 4, t = tid & 15;
    float row[TN];
    for (int u = 0; u < TN; ++u) row[u] = sc_s[b][t][u];
    for (int kk = 0; kk < KW; ++kk) {
      int best = 0; float bvv = row[0];
      for (int u = 1; u < TN; ++u)
        if (row[u] > bvv) { bvv = row[u]; best = u; }
      rout[(b * TN + t) * KW + kk] = best;
      row[best] = -3e38f;
    }
  }
}

// ---------------------------------------------------------------------------
// Kernel 4: main fused kernel.  Block = 16 tokens x all 16 tiles, 8 waves.
// LN(16 tokens) -> bf16 LDS (padded, conflict-free b128 reads).  Per tile:
// issue residual x loads early (single base + immediate offsets), gather
// combined[16x320] A-fragments from LDS via routes, 80 WMMAs for GEMM1
// (+bias), exact GELU, re-fragment 16x32 h-chunks through per-wave LDS,
// 16 WMMAs for GEMM2 (+bias), residual add, batched stores.
// ---------------------------------------------------------------------------
__global__ __launch_bounds__(256) void wte_main(const float* __restrict__ x,
                                                const float* __restrict__ gamma,
                                                const float* __restrict__ beta,
                                                const __bf16* __restrict__ w1f,
                                                const __bf16* __restrict__ w2f,
                                                const float* __restrict__ b1,
                                                const float* __restrict__ b2,
                                                const int* __restrict__ rout,
                                                float* __restrict__ out) {
  __shared__ __bf16 xn_s[16][1032];   // 16 tokens x 1024 (+8 pad) bf16
  __shared__ __bf16 h_s[8][16][40];   // per-wave 16x32 GELU chunk (+8 pad)
  int tid = threadIdx.x, lane = tid & 31, wv = tid >> 5;
  int b  = blockIdx.x >> 8;
  int s0 = (blockIdx.x & 255) << 4;
  long base = ((long)b * SS + s0) * DN;

  // --- LayerNorm 16 tokens into LDS as bf16 (one token per wave pass) ---
  for (int t2 = wv; t2 < 16; t2 += 8) {
    const float4* xr = (const float4*)(x + base + (long)t2 * DN);
    float4 xs[8];
    float s = 0.f, s2 = 0.f;
#pragma unroll
    for (int i = 0; i < 8; ++i) {
      xs[i] = xr[lane + 32 * i];
      s  += xs[i].x + xs[i].y + xs[i].z + xs[i].w;
      s2 += xs[i].x * xs[i].x + xs[i].y * xs[i].y + xs[i].z * xs[i].z + xs[i].w * xs[i].w;
    }
#pragma unroll
    for (int off = 16; off > 0; off >>= 1) {
      s  += __shfl_xor(s, off, 32);
      s2 += __shfl_xor(s2, off, 32);
    }
    float mu = s * (1.0f / DN);
    float rs = rsqrtf(s2 * (1.0f / DN) - mu * mu + 1e-5f);
#pragma unroll
    for (int i = 0; i < 8; ++i) {
      int c4 = lane + 32 * i;
      float4 gv = ((const float4*)gamma)[c4];
      float4 bv = ((const float4*)beta)[c4];
      v4bf o;
      o[0] = (__bf16)((xs[i].x - mu) * rs * gv.x + bv.x);
      o[1] = (__bf16)((xs[i].y - mu) * rs * gv.y + bv.y);
      o[2] = (__bf16)((xs[i].z - mu) * rs * gv.z + bv.z);
      o[3] = (__bf16)((xs[i].w - mu) * rs * gv.w + bv.w);
      *(v4bf*)&xn_s[t2][c4 * 4] = o;
    }
  }
  __syncthreads();

  int m  = lane & 15;     // token row / N column within fragments
  int hw = lane >> 4;     // half-wave select
  const v16bf* w1v = (const v16bf*)w1f;
  const v16bf* w2v = (const v16bf*)w2f;

  for (int tt = wv; tt < TN; tt += 8) {
    int rts[KW];
#pragma unroll
    for (int j = 0; j < KW; ++j) rts[j] = rout[(b * TN + tt) * KW + j];

    // Per-lane base pointers for the residual epilogue: every element of this
    // tile's output lives at base + r*4096B + nt*64B immediate offsets.
    const float* xp = x   + base + (long)(hw * 8) * DN + tt * TDI + m;
    float*       op = out + base + (long)(hw * 8) * DN + tt * TDI + m;

    // Issue all 32 residual loads now; GEMM work below hides their latency.
    float rv[4][8];
#pragma unroll
    for (int nt = 0; nt < 4; ++nt)
#pragma unroll
      for (int r = 0; r < 8; ++r)
        rv[nt][r] = xp[r * DN + nt * 16];

    v8f acc2[4];
#pragma unroll
    for (int nt = 0; nt < 4; ++nt) {
      float bvv = b2[nt * 16 + m];
#pragma unroll
      for (int r = 0; r < 8; ++r) acc2[nt][r] = bvv;
    }

#pragma unroll
    for (int pp = 0; pp < 4; ++pp) {          // hidden K-block (32 cols each)
      v8f acc[2];
#pragma unroll
      for (int j = 0; j < 2; ++j) {
        float bvv = b1[(pp * 2 + j) * 16 + m];
#pragma unroll
        for (int r = 0; r < 8; ++r) acc[j][r] = bvv;
      }
#pragma unroll
      for (int kt = 0; kt < 10; ++kt) {       // combined-feature K step of 32
        int f0 = kt * 32 + hw * 16;           // A elems 0..7
        int f1 = f0 + 8;                      // A elems 8..15
        int sec0 = f0 >> 6, sec1 = f1 >> 6;   // constant per kt after unroll
        int c0 = ((sec0 ? rts[sec0 - 1] : tt) << 6) + (f0 & 63);
        int c1 = ((sec1 ? rts[sec1 - 1] : tt) << 6) + (f1 & 63);
        v16bf a = cat16(*(const v8bf*)&xn_s[m][c0], *(const v8bf*)&xn_s[m][c1]);
#pragma unroll
        for (int j = 0; j < 2; ++j) {
          v16bf wb = w1v[(kt * 8 + pp * 2 + j) * 32 + lane];
          acc[j] = __builtin_amdgcn_wmma_f32_16x16x32_bf16(
              false, a, false, wb, (short)0, acc[j], false, false);
        }
      }
      // exact GELU, D-layout -> A-layout via per-wave LDS chunk
#pragma unroll
      for (int j = 0; j < 2; ++j) {
#pragma unroll
        for (int r = 0; r < 8; ++r) {
          float v = acc[j][r];
          float ge = 0.5f * v * (1.0f + erff(v * 0.70710678118654752f));
          h_s[wv][r + hw * 8][j * 16 + m] = (__bf16)ge;
        }
      }
      v16bf a2 = cat16(*(const v8bf*)&h_s[wv][m][hw * 16],
                       *(const v8bf*)&h_s[wv][m][hw * 16 + 8]);
#pragma unroll
      for (int nt = 0; nt < 4; ++nt) {
        v16bf wb = w2v[(pp * 4 + nt) * 32 + lane];
        acc2[nt] = __builtin_amdgcn_wmma_f32_16x16x32_bf16(
            false, a2, false, wb, (short)0, acc2[nt], false, false);
      }
    }
    // residual add + batched stores (immediate offsets off one base address)
#pragma unroll
    for (int nt = 0; nt < 4; ++nt)
#pragma unroll
      for (int r = 0; r < 8; ++r)
        op[r * DN + nt * 16] = rv[nt][r] + acc2[nt][r];
  }
}

// ---------------------------------------------------------------------------
// Launch
// ---------------------------------------------------------------------------
static const size_t OFF_PART = 0;                                   // 512*1024 f32
static const size_t OFF_W1F  = OFF_PART + (size_t)512 * 1024 * 4;   // 2 MB
static const size_t OFF_W2F  = OFF_W1F + (size_t)10 * 8 * 32 * 16 * 2;
static const size_t OFF_RT   = OFF_W2F + (size_t)4 * 4 * 32 * 16 * 2;

extern "C" void kernel_launch(void* const* d_in, const int* in_sizes, int n_in,
                              void* d_out, int out_size, void* d_ws, size_t ws_size,
                              hipStream_t stream) {
  const float* x     = (const float*)d_in[0];
  const float* gamma = (const float*)d_in[1];
  const float* beta  = (const float*)d_in[2];
  const float* Wq    = (const float*)d_in[3];
  const float* bq    = (const float*)d_in[4];
  const float* Wk    = (const float*)d_in[5];
  const float* bk    = (const float*)d_in[6];
  const float* W1    = (const float*)d_in[7];
  const float* b1    = (const float*)d_in[8];
  const float* W2    = (const float*)d_in[9];
  const float* b2    = (const float*)d_in[10];
  float* out = (float*)d_out;
  char* ws = (char*)d_ws;
  float*  part = (float*)(ws + OFF_PART);
  __bf16* w1f  = (__bf16*)(ws + OFF_W1F);
  __bf16* w2f  = (__bf16*)(ws + OFF_W2F);
  int*    rts  = (int*)(ws + OFF_RT);

  wte_prep <<<192, 256, 0, stream>>>(W1, W2, w1f, w2f);
  wte_stats<<<512, 256, 0, stream>>>(x, gamma, beta, part);
  wte_route<<<1,   256, 0, stream>>>(part, Wq, bq, Wk, bk, rts);
  wte_main <<<BB * (SS / 16), 256, 0, stream>>>(x, gamma, beta, w1f, w2f, b1, b2, rts, out);

  (void)in_sizes; (void)n_in; (void)out_size; (void)ws_size;
}
